// VectorQuantizer_45114336477231
// MI455X (gfx1250) — matrix operand order repaired and checked by reading, not compiled, not run on hardware
//
#include <hip/hip_runtime.h>
#include <math.h>

// ---------------------------------------------------------------------------
// VQ-VAE vector quantizer for MI455X (gfx1250, wave32, WMMA).
//   inputs : [65536, 256] f32     emb_w : [1024, 256] f32
//   d_out  : quantized [65536*256] f32  ++  loss (1)  ++  perplexity (1)
// Distance matmul runs on v_wmma_f32_16x16x32_bf16 (f32 accumulate); the
// ||e||^2 term, the argmin bookkeeping, the gather, and the loss are full f32.
// ---------------------------------------------------------------------------

#define N_ROWS   65536
#define DIM      256
#define K_CODES  1024
#define ROWS_PER_BLOCK 128   // 8 waves x 16 rows
#define B_STRIDE_U16   264   // 528 bytes per staged code row (bank spreading)

typedef __attribute__((ext_vector_type(16))) __bf16 v16bf;
typedef __attribute__((ext_vector_type(8)))  float  v8f;
typedef __attribute__((ext_vector_type(4)))  float  v4f;   // native vector (NT stores)

union BfFrag {
    v16bf v;
    unsigned short u[16];
    uint4 q[2];
};

__device__ __forceinline__ unsigned short f32_to_bf16_rne(float f) {
    unsigned int u = __float_as_uint(f);
    u += 0x7FFFu + ((u >> 16) & 1u);       // round-to-nearest-even
    return (unsigned short)(u >> 16);
}

// ---------------------------------------------------------------------------
// Kernel 1: codebook f32 -> bf16, ||e||^2 per code, zero histogram/partials.
// One wave per code row; lane handles 8 contiguous dims.
// ---------------------------------------------------------------------------
__global__ void vq_prep(const float* __restrict__ emb,
                        unsigned short* __restrict__ emb16,
                        float* __restrict__ enorm,
                        unsigned int* __restrict__ zero_words) {
    const int lane = threadIdx.x & 31;
    const int wave = threadIdx.x >> 5;
    const int row  = blockIdx.x * 8 + wave;

    const int gtid = blockIdx.x * blockDim.x + threadIdx.x;
    if (gtid < 1536) zero_words[gtid] = 0u;   // counts[1024] + partial_sse[512]

    const float* er = emb + row * DIM + lane * 8;
    v4f a = *(const v4f*)(er);
    v4f b = *(const v4f*)(er + 4);

    float ss = a.x*a.x + a.y*a.y + a.z*a.z + a.w*a.w
             + b.x*b.x + b.y*b.y + b.z*b.z + b.w*b.w;

    unsigned int w0 = (unsigned)f32_to_bf16_rne(a.x) | ((unsigned)f32_to_bf16_rne(a.y) << 16);
    unsigned int w1 = (unsigned)f32_to_bf16_rne(a.z) | ((unsigned)f32_to_bf16_rne(a.w) << 16);
    unsigned int w2 = (unsigned)f32_to_bf16_rne(b.x) | ((unsigned)f32_to_bf16_rne(b.y) << 16);
    unsigned int w3 = (unsigned)f32_to_bf16_rne(b.z) | ((unsigned)f32_to_bf16_rne(b.w) << 16);
    uint4 pk; pk.x = w0; pk.y = w1; pk.z = w2; pk.w = w3;
    *(uint4*)(emb16 + row * DIM + lane * 8) = pk;

    #pragma unroll
    for (int m = 16; m >= 1; m >>= 1) ss += __shfl_xor(ss, m, 32);
    if (lane == 0) enorm[row] = ss;
}

// ---------------------------------------------------------------------------
// Kernel 2: distances via WMMA, argmin, histogram, gather, partial SSE.
// ---------------------------------------------------------------------------
__global__ void __launch_bounds__(256)
vq_main(const float* __restrict__ inp,
        const float* __restrict__ emb,
        const unsigned short* __restrict__ emb16,
        const float* __restrict__ enorm,
        unsigned int* __restrict__ counts,
        float* __restrict__ partial_sse,
        float* __restrict__ out) {
    __shared__ unsigned short btile[2][16 * B_STRIDE_U16];  // double-buffered B tile
    __shared__ float enorm_s[K_CODES];
    __shared__ int   idx_s[8][16];
    __shared__ float sse_s[8];

    const int tid  = threadIdx.x;
    const int lane = tid & 31;
    const int wave = tid >> 5;
    const int hi   = lane >> 4;     // half-wave select (ISA K-split)
    const int col  = lane & 15;     // B column / C column / A row
    const int rowBase = blockIdx.x * ROWS_PER_BLOCK + wave * 16;

    // stage ||e||^2 for the whole codebook (4 KB)
    enorm_s[tid]       = enorm[tid];
    enorm_s[tid + 256] = enorm[tid + 256];
    enorm_s[tid + 512] = enorm[tid + 512];
    enorm_s[tid + 768] = enorm[tid + 768];

    // ---- A fragments: 16 rows x 256 dims, bf16, contiguous K split (same order as B) ----
    // lanes 0-15 : elems 0-15 -> K = c*32+0..15
    // lanes16-31 : elems 0-15 -> K = c*32+16..31
    v16bf afrag[8];
    {
        const float* xr = inp + (rowBase + col) * DIM;
        #pragma unroll
        for (int c = 0; c < 8; ++c) {
            const int kb = c * 32 + hi * 16;
            v4f p0 = *(const v4f*)(xr + kb);
            v4f p1 = *(const v4f*)(xr + kb + 4);
            v4f p2 = *(const v4f*)(xr + kb + 8);
            v4f p3 = *(const v4f*)(xr + kb + 12);
            BfFrag f;
            f.u[0]  = f32_to_bf16_rne(p0.x); f.u[1]  = f32_to_bf16_rne(p0.y);
            f.u[2]  = f32_to_bf16_rne(p0.z); f.u[3]  = f32_to_bf16_rne(p0.w);
            f.u[4]  = f32_to_bf16_rne(p1.x); f.u[5]  = f32_to_bf16_rne(p1.y);
            f.u[6]  = f32_to_bf16_rne(p1.z); f.u[7]  = f32_to_bf16_rne(p1.w);
            f.u[8]  = f32_to_bf16_rne(p2.x); f.u[9]  = f32_to_bf16_rne(p2.y);
            f.u[10] = f32_to_bf16_rne(p2.z); f.u[11] = f32_to_bf16_rne(p2.w);
            f.u[12] = f32_to_bf16_rne(p3.x); f.u[13] = f32_to_bf16_rne(p3.y);
            f.u[14] = f32_to_bf16_rne(p3.z); f.u[15] = f32_to_bf16_rne(p3.w);
            afrag[c] = f.v;
        }
    }

    float minval[8];
    int   minidx[8];
    #pragma unroll
    for (int v = 0; v < 8; ++v) { minval[v] = 3.4e38f; minidx[v] = 0; }

    // stage B tile 0 (16 codes x 256 bf16 -> 8 KB); each thread moves 32 B
    {
        const int r = tid >> 4, seg = tid & 15;
        const uint4* src = (const uint4*)(emb16 + r * DIM + seg * 16);
        uint4* dst = (uint4*)(&btile[0][r * B_STRIDE_U16 + seg * 16]);
        dst[0] = src[0]; dst[1] = src[1];
    }
    __syncthreads();

    for (int nt = 0; nt < 64; ++nt) {
        const int cur = nt & 1;
        if (nt + 1 < 64) {   // prefetch next tile into the other buffer
            const int r = tid >> 4, seg = tid & 15;
            const uint4* src = (const uint4*)(emb16 + ((nt + 1) * 16 + r) * DIM + seg * 16);
            uint4* dst = (uint4*)(&btile[cur ^ 1][r * B_STRIDE_U16 + seg * 16]);
            dst[0] = src[0]; dst[1] = src[1];
        }

        // B fragment per lane: 16 consecutive K bf16 of code `col`
        // lanes 0-15 -> K = c*32+0..15, lanes 16-31 -> K = c*32+16..31
        v8f acc = {0.f, 0.f, 0.f, 0.f, 0.f, 0.f, 0.f, 0.f};
        const unsigned short* bb = &btile[cur][col * B_STRIDE_U16 + hi * 16];
        #pragma unroll
        for (int c = 0; c < 8; ++c) {
            BfFrag bf;
            const uint4* bp = (const uint4*)(bb + c * 32);
            bf.q[0] = bp[0]; bf.q[1] = bp[1];
            acc = __builtin_amdgcn_wmma_f32_16x16x32_bf16(
                      false, afrag[c], false, bf.v, (short)0, acc, false, false);
        }

        // distance = ||e||^2 - 2 x.e  (||x||^2 is row-constant, irrelevant to argmin)
        const int   g  = nt * 16 + col;
        const float en = enorm_s[g];
        #pragma unroll
        for (int v = 0; v < 8; ++v) {
            const float d = en - 2.0f * acc[v];
            if (d < minval[v]) { minval[v] = d; minidx[v] = g; }
        }
        __syncthreads();   // everyone done reading btile[cur] before it is restaged
    }

    // ---- cross-lane argmin. C layout: VGPR v -> row v (lanes 0-15) / v+8 (16-31)
    #pragma unroll
    for (int v = 0; v < 8; ++v) {
        float val = minval[v];
        int   idx = minidx[v];
        #pragma unroll
        for (int m = 8; m >= 1; m >>= 1) {   // masks < 16 keep half-waves separate
            float ov = __shfl_xor(val, m, 32);
            int   oi = __shfl_xor(idx, m, 32);
            if (ov < val || (ov == val && oi < idx)) { val = ov; idx = oi; }
        }
        if (col == 0) {                       // lane 0 -> row v, lane 16 -> row v+8
            idx_s[wave][hi * 8 + v] = idx;
            atomicAdd(&counts[idx], 1u);      // integer atomics: deterministic
        }
    }
    __syncthreads();

    // ---- gather winning codes (f32), stream out, accumulate SSE ----
    float sse = 0.f;
    for (int r = 0; r < 16; ++r) {
        const int g = idx_s[wave][r];
        const float* e = emb + g * DIM + lane * 8;
        const float* x = inp + (rowBase + r) * DIM + lane * 8;
        float*       o = out + (rowBase + r) * DIM + lane * 8;
        v4f e0 = *(const v4f*)(e);
        v4f e1 = *(const v4f*)(e + 4);
        v4f x0 = *(const v4f*)(x);
        v4f x1 = *(const v4f*)(x + 4);
        __builtin_nontemporal_store(e0, (v4f*)o);
        __builtin_nontemporal_store(e1, (v4f*)(o + 4));
        float d;
        d = e0.x - x0.x; sse += d * d;  d = e0.y - x0.y; sse += d * d;
        d = e0.z - x0.z; sse += d * d;  d = e0.w - x0.w; sse += d * d;
        d = e1.x - x1.x; sse += d * d;  d = e1.y - x1.y; sse += d * d;
        d = e1.z - x1.z; sse += d * d;  d = e1.w - x1.w; sse += d * d;
    }
    #pragma unroll
    for (int m = 16; m >= 1; m >>= 1) sse += __shfl_xor(sse, m, 32);
    if (lane == 0) sse_s[wave] = sse;
    __syncthreads();
    if (tid == 0) {
        float s = 0.f;
        #pragma unroll
        for (int w = 0; w < 8; ++w) s += sse_s[w];     // fixed order: deterministic
        partial_sse[blockIdx.x] = s;
    }
}

// ---------------------------------------------------------------------------
// Kernel 3: loss + perplexity (deterministic fixed-order tree sums).
// ---------------------------------------------------------------------------
__global__ void vq_finalize(const unsigned int* __restrict__ counts,
                            const float* __restrict__ partial_sse,
                            float* __restrict__ out_scalars) {
    __shared__ float red[1024];
    const int t = threadIdx.x;

    const float p = (float)counts[t] * (1.0f / 65536.0f);
    red[t] = p * logf(p + 1e-10f);
    __syncthreads();
    for (int s = 512; s > 0; s >>= 1) {
        if (t < s) red[t] += red[t + s];
        __syncthreads();
    }
    const float H = red[0];
    __syncthreads();

    red[t] = (t < 512) ? partial_sse[t] : 0.f;
    __syncthreads();
    for (int s = 512; s > 0; s >>= 1) {
        if (t < s) red[t] += red[t + s];
        __syncthreads();
    }
    if (t == 0) {
        const float mse = red[0] * (1.0f / 16777216.0f);
        out_scalars[0] = mse * 1.25f;    // q_latent + 0.25 * e_latent (equal values)
        out_scalars[1] = expf(-H);
    }
}

// ---------------------------------------------------------------------------
extern "C" void kernel_launch(void* const* d_in, const int* in_sizes, int n_in,
                              void* d_out, int out_size, void* d_ws, size_t ws_size,
                              hipStream_t stream) {
    const float* inp = (const float*)d_in[0];   // [65536, 256]
    const float* emb = (const float*)d_in[1];   // [1024, 256]
    float* out = (float*)d_out;                 // quantized ++ loss ++ perplexity

    char* ws = (char*)d_ws;
    unsigned short* emb16   = (unsigned short*)(ws);            // 524288 B
    float*          enorm   = (float*)(ws + 524288);            //   4096 B
    unsigned int*   counts  = (unsigned int*)(ws + 528384);     //   4096 B
    float*          partial = (float*)(ws + 532480);            //   2048 B
    // counts+partial are contiguous: vq_prep zeroes those 1536 words each call.

    vq_prep    <<<128, 256, 0, stream>>>(emb, emb16, enorm, counts);
    vq_main    <<<512, 256, 0, stream>>>(inp, emb, emb16, enorm, counts, partial, out);
    vq_finalize<<<1, 1024, 0, stream>>>(counts, partial, out + (size_t)N_ROWS * DIM);
}
